// MoELayer_71751723647790
// MI455X (gfx1250) — matrix operand order repaired and checked by reading, not compiled, not run on hardware
//
#include <hip/hip_runtime.h>
#include <hip/hip_bf16.h>
#include <math.h>

// ---------------------------------------------------------------------------
// MoE layer for MI455X (gfx1250, wave32, WMMA).
//   Dense all-expert compute (faithful to reference) in bf16 WMMA:
//   2.2 TFLOP vs ~1.5 GB HBM traffic -> compute bound -> v_wmma path.
//   128x128 block tiles (~65 FLOP/byte of LDS-staged operands), pre-converted
//   bf16 operands, double-buffered LDS fed by global_load_async_to_lds_b128.
// ---------------------------------------------------------------------------

typedef __attribute__((ext_vector_type(16))) __bf16 v16bf;
typedef __attribute__((ext_vector_type(8)))  float  v8f;
typedef int v4i __attribute__((ext_vector_type(4)));

#define TOKS 4096   // B*S
#define DIM  2048   // D
#define HID  4096   // H
#define NEXP 8

// ---- CDNA5 async global->LDS copy (ASYNCcnt), with safe fallback ----------
#if defined(__has_builtin)
#  if __has_builtin(__builtin_amdgcn_global_load_async_to_lds_b128)
#    define HAVE_ASYNC_LDS 1
#  else
#    define HAVE_ASYNC_LDS 0
#  endif
#  if __has_builtin(__builtin_amdgcn_s_wait_asynccnt)
#    define WAIT_ASYNC0() __builtin_amdgcn_s_wait_asynccnt(0)
#  else
#    define WAIT_ASYNC0() asm volatile("s_wait_asynccnt 0x0" ::: "memory")
#  endif
#else
#  define HAVE_ASYNC_LDS 0
#  define WAIT_ASYNC0() asm volatile("s_wait_asynccnt 0x0" ::: "memory")
#endif

// builtin signature (from hipcc diagnostic): (v4i AS1*, v4i AS3*, imm, imm)
typedef __attribute__((address_space(1))) v4i g_v4i;   // global int4
typedef __attribute__((address_space(3))) v4i l_v4i;   // LDS int4

// ---------------------------------------------------------------------------
// Gating: logits = x @ Wg + bg, softmax, top-2 -> dense comb weights [T][NE]
// ---------------------------------------------------------------------------
__global__ __launch_bounds__(256) void gate_kernel(
    const float* __restrict__ x, const float* __restrict__ Wg,
    const float* __restrict__ bg, float* __restrict__ comb)
{
    const int t = blockIdx.x;
    __shared__ __align__(16) float xs[DIM];
    __shared__ float logits[NEXP];

    const float* xr = x + (size_t)t * DIM;
    for (int i = threadIdx.x; i < DIM; i += 256) xs[i] = xr[i];
    __syncthreads();

    const int wave = threadIdx.x >> 5;
    const int lane = threadIdx.x & 31;
    float p = 0.0f;
    for (int d = lane; d < DIM; d += 32)
        p += xs[d] * Wg[d * NEXP + wave];
    #pragma unroll
    for (int off = 16; off; off >>= 1) p += __shfl_xor(p, off, 32);
    if (lane == 0) logits[wave] = p + bg[wave];
    __syncthreads();

    if (threadIdx.x == 0) {
        float l[NEXP], mx = -1e30f;
        #pragma unroll
        for (int i = 0; i < NEXP; ++i) { l[i] = logits[i]; mx = fmaxf(mx, l[i]); }
        float s = 0.0f;
        #pragma unroll
        for (int i = 0; i < NEXP; ++i) { l[i] = __expf(l[i] - mx); s += l[i]; }
        const float inv = 1.0f / s;
        #pragma unroll
        for (int i = 0; i < NEXP; ++i) l[i] *= inv;
        int i0 = 0;
        #pragma unroll
        for (int i = 1; i < NEXP; ++i) if (l[i] > l[i0]) i0 = i;
        int i1 = (i0 == 0) ? 1 : 0;
        #pragma unroll
        for (int i = 0; i < NEXP; ++i) if (i != i0 && l[i] > l[i1]) i1 = i;
        #pragma unroll
        for (int i = 0; i < NEXP; ++i)
            comb[(size_t)t * NEXP + i] = (i == i0) ? l[i0] : (i == i1) ? l[i1] : 0.0f;
    }
}

// ---------------------------------------------------------------------------
// Flat fp32 -> bf16 conversion (for x). n must be a multiple of 1024.
// ---------------------------------------------------------------------------
__global__ __launch_bounds__(256) void conv_bf16_kernel(
    const float* __restrict__ in, __bf16* __restrict__ out)
{
    const size_t i = ((size_t)blockIdx.x * 256 + threadIdx.x) * 4;
    const float4 v = *(const float4*)(in + i);
    union { __bf16 b[4]; uint2 u; } pk;
    pk.b[0] = (__bf16)v.x; pk.b[1] = (__bf16)v.y;
    pk.b[2] = (__bf16)v.z; pk.b[3] = (__bf16)v.w;
    *(uint2*)(out + i) = pk.u;
}

// ---------------------------------------------------------------------------
// Weight convert + transpose:  W[K][N] fp32  ->  WT[N][K] bf16
// 32x32 LDS-tiled, both sides coalesced.
// ---------------------------------------------------------------------------
__global__ __launch_bounds__(256) void conv_transpose_kernel(
    const float* __restrict__ W, __bf16* __restrict__ WT, int K, int N)
{
    __shared__ float tile[32][33];
    const int kBase = blockIdx.y * 32, nBase = blockIdx.x * 32;
    const int tx = threadIdx.x & 31, ty = threadIdx.x >> 5;   // 32 x 8
    #pragma unroll
    for (int i = 0; i < 32; i += 8)
        tile[ty + i][tx] = W[(size_t)(kBase + ty + i) * N + nBase + tx];
    __syncthreads();
    #pragma unroll
    for (int i = 0; i < 32; i += 8)
        WT[(size_t)(nBase + ty + i) * K + kBase + tx] = (__bf16)tile[tx][ty + i];
}

// ---------------------------------------------------------------------------
// WMMA GEMM:  C[M,N] = A[M,K] @ WT[N,K]^T (+bias), bf16 operands.
//   Block tile 128x128, KT=32, double-buffered LDS via async global->LDS.
//   8 waves in 2(M)x4(N); each wave: 64x32 tile = 4x2 v_wmma_f32_16x16x32_bf16.
// EPI==0: h = gelu(acc + bias) -> bf16            (GEMM1 / GEMM2)
// EPI==1: out (+)= comb[token][e] * (acc + bias)  (GEMM3, fp32 out)
// ---------------------------------------------------------------------------
template<int EPI>
__global__ __launch_bounds__(256) void moe_gemm(
    const __bf16* __restrict__ A,    // [M][K] bf16
    const __bf16* __restrict__ WT,   // [N][K] bf16 (pre-transposed)
    const float*  __restrict__ bias, // [N]
    __bf16*       __restrict__ Hout, // EPI==0
    float*        __restrict__ Out,  // EPI==1
    const float*  __restrict__ comb, // [M][NE]
    int M, int N, int K, int expert, int accumulate)
{
    // [2 buffers][128 rows][40 bf16] ; row stride 80 B keeps 16 B alignment
    __shared__ __align__(16) __bf16 lA[2 * 128 * 40];
    __shared__ __align__(16) __bf16 lB[2 * 128 * 40];

    const int tid     = threadIdx.x;
    const int rowBase = blockIdx.y * 128;
    const int colBase = blockIdx.x * 128;

    const int wid   = tid >> 5;
    const int lane  = tid & 31;
    const int wm    = wid >> 2;     // 0..1 : 64-row band
    const int wn    = wid & 3;      // 0..3 : 32-col band
    const int lrow  = lane & 15;
    const int khalf = lane >> 4;

    v8f acc[4][2] = {};

    // Stage one 128x32 bf16 tile for A and B into LDS buffer `buf`.
    auto stage = [&](int buf, int k0) {
        const int base = buf * 5120;
        #pragma unroll
        for (int i = 0; i < 2; ++i) {               // 512 x 16B chunks each
            const int f = i * 256 + tid;
            const int r = f >> 2, c = f & 3;        // row, 8-bf16 chunk
            const __bf16* ga = A  + (size_t)(rowBase + r) * K + k0 + c * 8;
            const __bf16* gb = WT + (size_t)(colBase + r) * K + k0 + c * 8;
            __bf16* la = &lA[base + r * 40 + c * 8];
            __bf16* lb = &lB[base + r * 40 + c * 8];
#if HAVE_ASYNC_LDS
            __builtin_amdgcn_global_load_async_to_lds_b128(
                (g_v4i*)ga, (l_v4i*)la, 0, 0);
            __builtin_amdgcn_global_load_async_to_lds_b128(
                (g_v4i*)gb, (l_v4i*)lb, 0, 0);
#else
            *(uint4*)la = *(const uint4*)ga;
            *(uint4*)lb = *(const uint4*)gb;
#endif
        }
    };

    const int nk = K >> 5;
    stage(0, 0);

    for (int ks = 0; ks < nk; ++ks) {
        WAIT_ASYNC0();        // my async copies into current buffer done
        __syncthreads();      // everyone's copies done; prev compute done
        if (ks + 1 < nk) stage((ks + 1) & 1, (ks + 1) << 5);

        const int base = (ks & 1) * 5120;
        union F { v16bf v; uint4 q[2]; };
        F fa[4], fb[2];
        #pragma unroll
        for (int mi = 0; mi < 4; ++mi) {
            // A 16x32: lane<16 -> K 0..7 & 16..23 ; lane>=16 -> K 8..15 & 24..31
            const __bf16* ar = &lA[base + (wm * 64 + mi * 16 + lrow) * 40];
            fa[mi].q[0] = *(const uint4*)(ar + khalf * 8);
            fa[mi].q[1] = *(const uint4*)(ar + 16 + khalf * 8);
        }
        #pragma unroll
        for (int ni = 0; ni < 2; ++ni) {
            // B 32x16: lane<16 -> col=lane, K 0..7 & 16..23 ; lane>=16 -> K 8..15 & 24..31
            const __bf16* br = &lB[base + (wn * 32 + ni * 16 + lrow) * 40];
            fb[ni].q[0] = *(const uint4*)(br + khalf * 8);
            fb[ni].q[1] = *(const uint4*)(br + 16 + khalf * 8);
        }
        #pragma unroll
        for (int mi = 0; mi < 4; ++mi)
            #pragma unroll
            for (int ni = 0; ni < 2; ++ni)
                acc[mi][ni] = __builtin_amdgcn_wmma_f32_16x16x32_bf16(
                    false, fa[mi].v, false, fb[ni].v,
                    (short)0, acc[mi][ni], false, false);
    }

    // ---- epilogue ----
    #pragma unroll
    for (int mi = 0; mi < 4; ++mi) {
        #pragma unroll
        for (int ni = 0; ni < 2; ++ni) {
            const int col = colBase + wn * 32 + ni * 16 + lrow;
            const float bv = bias[col];
            #pragma unroll
            for (int r = 0; r < 8; ++r) {
                // C/D: VGPR r -> M=r (lanes 0-15) or M=r+8 (lanes 16-31)
                const int row = rowBase + wm * 64 + mi * 16 + r + (khalf ? 8 : 0);
                const float val = acc[mi][ni][r] + bv;
                if (EPI == 0) {
                    const float g = 0.5f * val * (1.0f + erff(val * 0.70710678118654752f));
                    Hout[(size_t)row * N + col] = (__bf16)g;
                } else {
                    const float w = comb[(size_t)row * NEXP + expert];
                    float* o = &Out[(size_t)row * N + col];
                    if (accumulate) { if (w != 0.0f) *o += w * val; }
                    else            { *o = w * val; }   // expert 0 initializes out
                }
            }
        }
    }
}

// ---------------------------------------------------------------------------
extern "C" void kernel_launch(void* const* d_in, const int* in_sizes, int n_in,
                              void* d_out, int out_size, void* d_ws, size_t ws_size,
                              hipStream_t stream)
{
    const float* x  = (const float*)d_in[0];
    const float* W1 = (const float*)d_in[1];
    const float* b1 = (const float*)d_in[2];
    const float* W2 = (const float*)d_in[3];
    const float* b2 = (const float*)d_in[4];
    const float* W3 = (const float*)d_in[5];
    const float* b3 = (const float*)d_in[6];
    const float* Wg = (const float*)d_in[7];
    const float* bg = (const float*)d_in[8];
    float* out = (float*)d_out;

    // ws: comb[T][NE] f32 | xb[T][D] bf16 | h1[T][H] bf16 | h2[T][H] bf16 | wbT (<=H*H bf16)
    char* ws = (char*)d_ws;
    float*  comb = (float*)ws;
    size_t  off  = ((size_t)TOKS * NEXP * sizeof(float) + 255) & ~(size_t)255;
    __bf16* xb = (__bf16*)(ws + off);  off += (size_t)TOKS * DIM * sizeof(__bf16);
    __bf16* h1 = (__bf16*)(ws + off);  off += (size_t)TOKS * HID * sizeof(__bf16);
    __bf16* h2 = (__bf16*)(ws + off);  off += (size_t)TOKS * HID * sizeof(__bf16);
    __bf16* wbT = (__bf16*)(ws + off);

    gate_kernel<<<TOKS, 256, 0, stream>>>(x, Wg, bg, comb);
    conv_bf16_kernel<<<(TOKS * DIM / 4) / 256, 256, 0, stream>>>(x, xb);

    const dim3 blk(256);
    const dim3 gH(HID / 128, TOKS / 128);
    const dim3 gD(DIM / 128, TOKS / 128);

    for (int e = 0; e < NEXP; ++e) {
        // h1 = gelu(x @ W1_e + b1_e)          [T,D] x [D,H]
        conv_transpose_kernel<<<dim3(HID / 32, DIM / 32), blk, 0, stream>>>(
            W1 + (size_t)e * DIM * HID, wbT, DIM, HID);
        moe_gemm<0><<<gH, blk, 0, stream>>>(
            xb, wbT, b1 + (size_t)e * HID, h1, nullptr, nullptr,
            TOKS, HID, DIM, e, 0);
        // h2 = gelu(h1 @ W2_e + b2_e)         [T,H] x [H,H]
        conv_transpose_kernel<<<dim3(HID / 32, HID / 32), blk, 0, stream>>>(
            W2 + (size_t)e * HID * HID, wbT, HID, HID);
        moe_gemm<0><<<gH, blk, 0, stream>>>(
            h1, wbT, b2 + (size_t)e * HID, h2, nullptr, nullptr,
            TOKS, HID, HID, e, 0);
        // out (+)= comb[:,e] * (h2 @ W3_e + b3_e)   [T,H] x [H,D]
        conv_transpose_kernel<<<dim3(DIM / 32, HID / 32), blk, 0, stream>>>(
            W3 + (size_t)e * HID * DIM, wbT, HID, DIM);
        moe_gemm<1><<<gD, blk, 0, stream>>>(
            h2, wbT, b3 + (size_t)e * DIM, nullptr, out, comb,
            TOKS, DIM, HID, e, (e > 0) ? 1 : 0);
    }
}
